// MHA_65412351918121
// MI455X (gfx1250) — matrix operand order repaired and checked by reading, not compiled, not run on hardware
//
#include <hip/hip_runtime.h>

// ---------------------------------------------------------------------------
// MHA forward for MI455X (gfx1250): bf16 WMMA pipeline with flash attention.
//   B=4, S=2048, D=1024, H=16, hd=64
//   - async global->LDS staging (GLOBAL_LOAD_ASYNC_TO_LDS_B128), double-buffered
//   - V written pre-transposed by the QKV GEMM so flash needs no transposes
// ---------------------------------------------------------------------------

typedef __attribute__((ext_vector_type(16))) __bf16 bf16x16;
typedef __attribute__((ext_vector_type(8)))  __bf16 bf16x8;
typedef __attribute__((ext_vector_type(8)))  float  f32x8;
typedef __attribute__((ext_vector_type(4)))  float  f32x4;
typedef __attribute__((ext_vector_type(4)))  int    i32x4;

__device__ __forceinline__ bf16x16 ld_frag(const __bf16* lo_p, const __bf16* hi_p) {
    bf16x8 lo = *(const bf16x8*)lo_p;
    bf16x8 hi = *(const bf16x8*)hi_p;
    return __builtin_shufflevector(lo, hi, 0,1,2,3,4,5,6,7,8,9,10,11,12,13,14,15);
}

__device__ __forceinline__ f32x8 wmma_bf16(bf16x16 a, bf16x16 b, f32x8 c) {
    return __builtin_amdgcn_wmma_f32_16x16x32_bf16(
        /*neg_a=*/false, a, /*neg_b=*/false, b,
        /*c_mod=*/(short)0, c, /*reuse_a=*/false, /*reuse_b=*/false);
}

// ---- CDNA5 async global->LDS staging (guarded; falls back cleanly) ---------
#if defined(__has_builtin)
#if __has_builtin(__builtin_amdgcn_global_load_async_to_lds_b128)
#define ATHENA_ASYNC_LDS 1
#endif
#endif

#if defined(ATHENA_ASYNC_LDS)
__device__ __forceinline__ void stage16(const __bf16* g, __bf16* l) {
    __builtin_amdgcn_global_load_async_to_lds_b128(
        (__attribute__((address_space(1))) i32x4*)g,
        (__attribute__((address_space(3))) i32x4*)l, 0, 0);
}
__device__ __forceinline__ void stage_wait() {
#if __has_builtin(__builtin_amdgcn_s_wait_asynccnt)
    __builtin_amdgcn_s_wait_asynccnt(0);
#else
    asm volatile("s_wait_asynccnt 0x0" ::: "memory");
#endif
}
#else
__device__ __forceinline__ void stage16(const __bf16* g, __bf16* l) {
    *(bf16x8*)l = *(const bf16x8*)g;
}
__device__ __forceinline__ void stage_wait() {}
#endif

// ---------------------------------------------------------------------------
// Kernel 1: fp32 -> bf16 conversion (element counts are multiples of 8)
// ---------------------------------------------------------------------------
__global__ __launch_bounds__(256)
void mha_cvt_f32_bf16(const float* __restrict__ src, __bf16* __restrict__ dst, int n) {
    int i = (blockIdx.x * 256 + threadIdx.x) * 8;
    if (i + 8 <= n) {
        f32x4 a = *(const f32x4*)(src + i);
        f32x4 b = *(const f32x4*)(src + i + 4);
        bf16x8 o;
        #pragma unroll
        for (int j = 0; j < 4; ++j) { o[j] = (__bf16)a[j]; o[4 + j] = (__bf16)b[j]; }
        *(bf16x8*)(dst + i) = o;
    }
}

// ---------------------------------------------------------------------------
// Kernel 2/4: tiled bf16 GEMM  C[M,N] = A[M,K] * Wt[N,K]^T + bias
//   BM=128, BN=128, BK=32; 256 threads = 8 waves in a 4x2 grid,
//   each wave computes a 32x64 region = 2x4 WMMA tiles. Double-buffered async
//   LDS staging. QKV epilogue scatters Q,K as [B,H,S,64] and V as [B,H,64,S].
// ---------------------------------------------------------------------------
#define GEMM_BM 128
#define GEMM_BN 128
#define GEMM_BK 32
#define LDS_KS 40   // bf16 elements per row (32 + 8 pad, keeps 16B alignment)

template<bool QKV_SCATTER>
__global__ __launch_bounds__(256)
void mha_gemm_bf16(const __bf16* __restrict__ A,   // [M,K]
                   const __bf16* __restrict__ Wt,  // [N,K]
                   const float*  __restrict__ bias,// [N]
                   float* __restrict__ Cout,       // [M,N] (non-scatter)
                   __bf16* __restrict__ qb,
                   __bf16* __restrict__ kb,
                   __bf16* __restrict__ vb,        // [B,H,64,S] (transposed!)
                   int M, int N, int K) {
    __shared__ __align__(16) __bf16 lA[2][GEMM_BM * LDS_KS];
    __shared__ __align__(16) __bf16 lB[2][GEMM_BN * LDS_KS];

    const int tid  = threadIdx.x;
    const int wave = tid >> 5;
    const int lane = tid & 31;
    const int half = lane >> 4;
    const int l16  = lane & 15;
    const int wm   = wave >> 1;   // 0..3 : 32-row band
    const int wn   = wave & 1;    // 0..1 : 64-col band
    const int m0   = blockIdx.x * GEMM_BM;
    const int n0   = blockIdx.y * GEMM_BN;

    const int sr = tid >> 1;          // 0..127
    const int sc = (tid & 1) * 16;    // 0, 16

    auto issue = [&](int buf, int k0) {
        stage16(&A [(size_t)(m0 + sr) * K + k0 + sc],     &lA[buf][(size_t)sr * LDS_KS + sc]);
        stage16(&A [(size_t)(m0 + sr) * K + k0 + sc + 8], &lA[buf][(size_t)sr * LDS_KS + sc + 8]);
        stage16(&Wt[(size_t)(n0 + sr) * K + k0 + sc],     &lB[buf][(size_t)sr * LDS_KS + sc]);
        stage16(&Wt[(size_t)(n0 + sr) * K + k0 + sc + 8], &lB[buf][(size_t)sr * LDS_KS + sc + 8]);
    };

    f32x8 acc[2][4] = {};
    issue(0, 0);

    for (int k0 = 0; k0 < K; k0 += GEMM_BK) {
        const int buf = (k0 / GEMM_BK) & 1;
        stage_wait();
        __syncthreads();
        if (k0 + GEMM_BK < K) issue(buf ^ 1, k0 + GEMM_BK);

        // ---- A fragments (16x32, lane=row, K halves at half*8 / 16+half*8) ----
        bf16x16 af[2];
        #pragma unroll
        for (int mt = 0; mt < 2; ++mt) {
            const __bf16* ap = &lA[buf][(size_t)(wm * 32 + mt * 16 + l16) * LDS_KS];
            af[mt] = ld_frag(ap + half * 8, ap + 16 + half * 8);
        }
        // ---- B fragments (32x16, lane=col, K halves at half*8 / 16+half*8) ----
        bf16x16 bf[4];
        #pragma unroll
        for (int nt = 0; nt < 4; ++nt) {
            const __bf16* bp = &lB[buf][(size_t)(wn * 64 + nt * 16 + l16) * LDS_KS + half * 8];
            bf[nt] = ld_frag(bp, bp + 16);
        }
        #pragma unroll
        for (int mt = 0; mt < 2; ++mt)
            #pragma unroll
            for (int nt = 0; nt < 4; ++nt)
                acc[mt][nt] = wmma_bf16(af[mt], bf[nt], acc[mt][nt]);
    }

    // ---- epilogue: C layout row = i + half*8, col = l16 ----
    #pragma unroll
    for (int mt = 0; mt < 2; ++mt) {
        #pragma unroll
        for (int nt = 0; nt < 4; ++nt) {
            const int col = n0 + wn * 64 + nt * 16 + l16;
            const float bv = bias[col];
            const int row0 = m0 + wm * 32 + mt * 16 + half * 8;
            if (QKV_SCATTER) {
                // col = h*192 + r ; r<64 -> Q, r<128 -> K (natural), else V (transposed)
                const int hh = col / 192;
                const int r  = col % 192;
                const int bidx = row0 >> 11;       // / 2048
                const int s0   = row0 & 2047;      // i runs along s
                if (r >= 128) {
                    bf16x8 v8;
                    #pragma unroll
                    for (int i = 0; i < 8; ++i) v8[i] = (__bf16)(acc[mt][nt][i] + bv);
                    // V^T: [B,H,64,S]; s contiguous -> 16B store
                    *(bf16x8*)&vb[((((size_t)bidx * 16 + hh) * 64 + (r - 128)) << 11) + s0] = v8;
                } else {
                    __bf16* base = (r < 64) ? qb : kb;
                    #pragma unroll
                    for (int i = 0; i < 8; ++i)
                        base[((((size_t)bidx * 16 + hh) * 2048 + (s0 + i)) << 6) + (r & 63)] =
                            (__bf16)(acc[mt][nt][i] + bv);
                }
            } else {
                #pragma unroll
                for (int i = 0; i < 8; ++i)
                    Cout[(size_t)(row0 + i) * N + col] = acc[mt][nt][i] + bv;
            }
        }
    }
}

// ---------------------------------------------------------------------------
// Kernel 3: flash attention with TRANSPOSED logits (S^T = K Q^T).
//   One WG (128 thr = 4 waves) per (b*h, 64 query rows); each wave owns 16
//   queries -> one query per WMMA column lane. Softmax is in-lane VALU + one
//   xor-16 shuffle. V arrives pre-transposed, so both sK and sVt stage via
//   async 16B chunks (double-buffered). 16 WMMAs per 64-key block.
// ---------------------------------------------------------------------------
__global__ __launch_bounds__(128)
void mha_flash_attn(const __bf16* __restrict__ Q,   // [B,H,S,64]
                    const __bf16* __restrict__ K,   // [B,H,S,64]
                    const __bf16* __restrict__ V,   // [B,H,64,S]  (V^T)
                    __bf16* __restrict__ O) {       // [B,S,H*64]
    const int S = 2048, HD = 64;
    const int bh = blockIdx.y;            // b*16 + h
    const int b  = bh >> 4;
    const int h  = bh & 15;
    const int q0 = blockIdx.x * 64;

    const __bf16* Qp  = Q + (size_t)bh * S * HD;
    const __bf16* Kp  = K + (size_t)bh * S * HD;
    const __bf16* Vtp = V + (size_t)bh * S * HD;   // rows = hd (stride S)

    const int tid  = threadIdx.x;
    const int wave = tid >> 5;
    const int lane = tid & 31;
    const int half = lane >> 4;
    const int l16  = lane & 15;

    __shared__ __align__(16) __bf16 sK [2][64 * 72];   // [key][hd], stride 72
    __shared__ __align__(16) __bf16 sVt[2][64 * 72];   // [hd][key], stride 72
    __shared__ __align__(16) __bf16 sP [4][16 * 72];   // per-wave P^T [query][key]

    auto issue = [&](int buf, int kb) {
        #pragma unroll
        for (int l = 0; l < 4; ++l) {
            const int f = l * 128 + tid;            // 0..511 chunks of 8 elems
            const int r = f >> 3, c = (f & 7) * 8;  // r: key (sK) / hd (sVt)
            stage16(&Kp[(size_t)(kb + r) * HD + c],  &sK [buf][(size_t)r * 72 + c]);
            stage16(&Vtp[(size_t)r * S + kb + c],    &sVt[buf][(size_t)r * 72 + c]);
        }
    };

    // ---- Q^T B-fragments: lane = query col (l16), K(hd) halves at ks*32+half*8 / ks*32+16+half*8 ----
    // scale (1/sqrt(64)) and log2(e) folded into Q once: softmax uses exp2.
    bf16x16 qf[2];
    {
        const float qscale = 0.125f * 1.44269504088896340736f;
        const __bf16* qrow = Qp + (size_t)(q0 + wave * 16 + l16) * HD;
        #pragma unroll
        for (int ks = 0; ks < 2; ++ks) {
            bf16x16 t = ld_frag(qrow + ks * 32 + half * 8, qrow + ks * 32 + 16 + half * 8);
            #pragma unroll
            for (int j = 0; j < 16; ++j) qf[ks][j] = (__bf16)((float)t[j] * qscale);
        }
    }

    f32x8 oacc[4] = {};            // O^T: 4 hd tiles, col = this lane's query
    float m_run = -3.0e38f;        // per-lane (per-query) running max (log2 domain)
    float l_run = 0.0f;            // per-lane running sum

    issue(0, 0);

    for (int kb = 0; kb < S; kb += 64) {
        const int buf = (kb >> 6) & 1;
        stage_wait();
        __syncthreads();
        if (kb + 64 < S) issue(buf ^ 1, kb + 64);

        // ---- S^T = K Q^T : 4 key tiles (mt) x 2 hd k-steps ----
        f32x8 st[4];
        #pragma unroll
        for (int mt = 0; mt < 4; ++mt) {
            f32x8 c = {};
            #pragma unroll
            for (int ks = 0; ks < 2; ++ks) {
                const __bf16* ap = &sK[buf][(size_t)(mt * 16 + l16) * 72 + ks * 32];
                c = wmma_bf16(ld_frag(ap + half * 8, ap + 16 + half * 8), qf[ks], c);
            }
            st[mt] = c;
        }

        // ---- per-lane online softmax over this block's 32 logits ----
        float bm = st[0][0];
        #pragma unroll
        for (int mt = 0; mt < 4; ++mt)
            #pragma unroll
            for (int i = 0; i < 8; ++i) bm = fmaxf(bm, st[mt][i]);
        bm = fmaxf(bm, __shfl_xor(bm, 16, 32));          // merge the two halves
        const float m_new = fmaxf(m_run, bm);
        const float corr  = exp2f(m_run - m_new);
        m_run = m_new;

        float rs = 0.0f;
        __bf16* pw = sP[wave];
        #pragma unroll
        for (int mt = 0; mt < 4; ++mt) {
            bf16x8 p8;
            #pragma unroll
            for (int i = 0; i < 8; ++i) {
                const float p = exp2f(st[mt][i] - m_new);
                rs += p;
                p8[i] = (__bf16)p;
            }
            // P^T stored as [query][key]: key = mt*16 + half*8 + i (i contiguous)
            *(bf16x8*)&pw[(size_t)l16 * 72 + mt * 16 + half * 8] = p8;
        }
        rs += __shfl_xor(rs, 16, 32);                    // disjoint halves: full sum
        l_run = l_run * corr + rs;

        // ---- rescale O^T, then O^T += V^T P^T (4 hd tiles x 2 key k-steps) ----
        #pragma unroll
        for (int t = 0; t < 4; ++t)
            #pragma unroll
            for (int i = 0; i < 8; ++i) oacc[t][i] *= corr;

        bf16x16 pf[2];
        #pragma unroll
        for (int ks = 0; ks < 2; ++ks) {
            const __bf16* pp = &pw[(size_t)l16 * 72 + ks * 32 + half * 8];
            pf[ks] = ld_frag(pp, pp + 16);
        }
        #pragma unroll
        for (int t = 0; t < 4; ++t) {
            #pragma unroll
            for (int ks = 0; ks < 2; ++ks) {
                const __bf16* ap = &sVt[buf][(size_t)(t * 16 + l16) * 72 + ks * 32];
                oacc[t] = wmma_bf16(ld_frag(ap + half * 8, ap + 16 + half * 8), pf[ks], oacc[t]);
            }
        }
    }

    // ---- normalize, store O[b, q, h*64 + dd]; i runs along dd -> 16B stores ----
    const float inv = 1.0f / l_run;
    const int   qg  = q0 + wave * 16 + l16;
    __bf16* op = O + ((size_t)(b * 2048 + qg) << 10) + h * 64;
    #pragma unroll
    for (int t = 0; t < 4; ++t) {
        bf16x8 o8;
        #pragma unroll
        for (int i = 0; i < 8; ++i) o8[i] = (__bf16)(oacc[t][i] * inv);
        *(bf16x8*)&op[t * 16 + half * 8] = o8;
    }
}

// ---------------------------------------------------------------------------
// Host launcher
// ---------------------------------------------------------------------------
extern "C" void kernel_launch(void* const* d_in, const int* in_sizes, int n_in,
                              void* d_out, int out_size, void* d_ws, size_t ws_size,
                              hipStream_t stream) {
    const float* x      = (const float*)d_in[0];   // [4,2048,1024]
    const float* qkv_w  = (const float*)d_in[1];   // [3072,1024]
    const float* qkv_b  = (const float*)d_in[2];   // [3072]
    const float* o_w    = (const float*)d_in[3];   // [1024,1024]
    const float* o_b    = (const float*)d_in[4];   // [1024]
    float* out = (float*)d_out;                    // [4,2048,1024]

    const size_t NX  = 8192u * 1024u;
    const size_t NQW = 3072u * 1024u;
    const size_t NOW = 1024u * 1024u;
    const size_t NQKV = 4u * 16u * 2048u * 64u;    // per Q/K/V buffer

    char* ws = (char*)d_ws;
    __bf16* xb   = (__bf16*)(ws);
    __bf16* wqkv = (__bf16*)(ws + NX * 2);
    __bf16* wo   = (__bf16*)(ws + (NX + NQW) * 2);
    __bf16* Qb   = (__bf16*)(ws + (NX + NQW + NOW) * 2);
    __bf16* Kb   = Qb + NQKV;
    __bf16* Vb   = Kb + NQKV;                      // stored transposed [B,H,64,S]
    __bf16* Ob   = Vb + NQKV;

    // 1) fp32 -> bf16 conversions
    mha_cvt_f32_bf16<<<dim3(NX  / 2048), dim3(256), 0, stream>>>(x, xb, (int)NX);
    mha_cvt_f32_bf16<<<dim3(NQW / 2048), dim3(256), 0, stream>>>(qkv_w, wqkv, (int)NQW);
    mha_cvt_f32_bf16<<<dim3(NOW / 2048), dim3(256), 0, stream>>>(o_w, wo, (int)NOW);

    // 2) QKV projection GEMM with scatter (Q,K natural; V transposed)
    mha_gemm_bf16<true><<<dim3(64, 24), dim3(256), 0, stream>>>(
        xb, wqkv, qkv_b, nullptr, Qb, Kb, Vb, 8192, 3072, 1024);

    // 3) flash attention -> Ob [B,S,1024] bf16
    mha_flash_attn<<<dim3(32, 64), dim3(128), 0, stream>>>(Qb, Kb, Vb, Ob);

    // 4) output projection -> fp32 out
    mha_gemm_bf16<false><<<dim3(64, 8), dim3(256), 0, stream>>>(
        Ob, wo, o_b, out, nullptr, nullptr, nullptr, 8192, 1024, 1024);
}
